// ContextAttention_927712936392
// MI455X (gfx1250) — matrix operand order repaired and checked by reading, not compiled, not run on hardware
//
#include <hip/hip_runtime.h>
#include <hip/hip_bf16.h>

// ---------------------------------------------------------------------------
// ContextAttention for MI455X (gfx1250): bf16 WMMA everywhere.
//   B=4, D=512, S=2048, L=8192
// ---------------------------------------------------------------------------

#define BB 4
#define DD 512
#define SS 2048
#define LL 8192
#define SCALE 0.044194173824159216f   // 1/sqrt(512)

typedef __attribute__((ext_vector_type(16))) __bf16 v16bf;
typedef __attribute__((ext_vector_type(8)))  __bf16 v8bf;
typedef __attribute__((ext_vector_type(8)))  float  v8f;

// A-matrix (16x32 bf16) lane layout, contiguous k (B frags use the same order):
//   lane m = lane&15 holds row m; kh = lane>>4 selects K halves
//   a[0:15] = K[kh*16 .. kh*16+15]
__device__ __forceinline__ v16bf a_frag_bf16(const __bf16* __restrict__ row,
                                             int kh) {
    v16bf a;
    ((v8bf*)&a)[0] = *(const v8bf*)(row + kh * 16);
    ((v8bf*)&a)[1] = *(const v8bf*)(row + kh * 16 + 8);
    return a;
}

__device__ __forceinline__ v8f wmma_bf16(v16bf a, v16bf b, v8f c) {
    return __builtin_amdgcn_wmma_f32_16x16x32_bf16(
        /*neg_a=*/false, a, /*neg_b=*/false, b,
        /*c_mod=*/(short)0, c, /*reuse_a=*/false, /*reuse_b=*/false);
}

// ---------------------------------------------------------------------------
// f32 -> bf16 bulk convert (vectorized x4)
// ---------------------------------------------------------------------------
__global__ __launch_bounds__(256) void cvt_f32_bf16_kernel(
    const float* __restrict__ in, __bf16* __restrict__ out, int n4) {
    int i = blockIdx.x * blockDim.x + threadIdx.x;
    if (i >= n4) return;
    float4 v = ((const float4*)in)[i];
    __bf16* o = out + i * 4;
    o[0] = (__bf16)v.x; o[1] = (__bf16)v.y;
    o[2] = (__bf16)v.z; o[3] = (__bf16)v.w;
}

// ---------------------------------------------------------------------------
// Projection: Out[b,o,s] = ELU( sum_d W[o,d] * H[b,d,s] + bias[o] ) -> bf16
//   grid: (D/64, S/16, B), block 128 (4 waves). Wave w does o-tile o0=bx*64+w*16.
//   H tile staged into LDS with gfx1250 async global->LDS copies (ASYNCcnt).
//   K path: ostride=1,  sstride=D   (K stored [B,S,D])
//   V path: ostride=S,  sstride=1   (V stored [B,D,S]  == transposed)
// ---------------------------------------------------------------------------
__global__ __launch_bounds__(128) void proj_elu_kernel(
    const float* __restrict__ W,    // [D][D]
    const float* __restrict__ bias, // [D]
    const float* __restrict__ H,    // [B][D][S]
    __bf16* __restrict__ Out,
    int ostride, int sstride) {
    __shared__ float Hf[32][16];    // raw f32 tile, filled by async DMA (2 KB)

    const int b     = blockIdx.z;
    const int s0    = blockIdx.y * 16;
    const int obase = blockIdx.x * 64;
    const int tid   = threadIdx.x;
    const int wave  = tid >> 5;
    const int lane  = tid & 31;
    const int m     = lane & 15;
    const int kh    = lane >> 4;
    const int o0    = obase + wave * 16;

    const float* __restrict__ Hb = H + (size_t)b * DD * SS;

    // per-thread async-copy source/dest for the [32 x 16] f32 tile:
    // thread t copies 16 bytes: row = t/4 (0..31), quarter = t%4
    const int trow = tid >> 2;
    const int tq   = tid & 3;
    const unsigned lds_off =
        (unsigned)(uintptr_t)(&Hf[trow][tq * 4]);   // low 32 bits == LDS offset

    v8f acc = {};
    for (int k0 = 0; k0 < DD; k0 += 32) {
        __syncthreads();                 // tile consumed by everyone
        {
            const float* gsrc = Hb + (size_t)(k0 + trow) * SS + s0 + tq * 4;
            asm volatile("global_load_async_to_lds_b128 %0, %1, off"
                         :: "v"(lds_off), "v"(gsrc)
                         : "memory");
            asm volatile("s_wait_asynccnt 0" ::: "memory");
        }
        __syncthreads();                 // tile visible to all waves

        // A frag: W rows (f32 -> bf16)
        const float* wrow = W + (size_t)(o0 + m) * DD + k0;
        v16bf a;
#pragma unroll
        for (int i = 0; i < 8; ++i) {
            a[i]     = (__bf16)wrow[kh * 8 + i];
            a[8 + i] = (__bf16)wrow[16 + kh * 8 + i];
        }
        // B frag: bf[k] = H[k0+kh*8+k][s0+n] (k<8), H[k0+8+kh*8+k][s0+n] (k>=8), lane n = m, via LDS
        v16bf bf;
#pragma unroll
        for (int k = 0; k < 16; ++k) bf[k] = (__bf16)Hf[(k < 8 ? 0 : 8) + kh * 8 + k][m];

        acc = wmma_bf16(a, bf, acc);
    }

    // epilogue: bias + ELU, scatter per C-layout
    __bf16* Ob = Out + (size_t)b * DD * SS;
#pragma unroll
    for (int r = 0; r < 8; ++r) {
        int mm = r + (kh << 3);
        int o  = o0 + mm;
        int s  = s0 + m;
        float x = acc[r] + bias[o];
        float e = x > 0.0f ? x : (__expf(x) - 1.0f);
        Ob[(size_t)o * ostride + (size_t)s * sstride] = (__bf16)e;
    }
}

// ---------------------------------------------------------------------------
// Fused attention pass:  E = scale*Q.K^T ; A = softmax(E) ; C = A.V
//   grid: (L/16, B), block 256 (8 waves). Block owns 16 query rows.
//   LDS: 64 KB bf16 buffer [16][S] holding scores, then probabilities.
// ---------------------------------------------------------------------------
__global__ __launch_bounds__(256) void attn_kernel(
    const __bf16* __restrict__ Q,   // [.][L][D] bf16, batch stride qbstride
    const __bf16* __restrict__ Kc,  // [B][S][D] bf16
    const __bf16* __restrict__ Vt,  // [B][D][S] bf16 (transposed V)
    float*  __restrict__ Cout,      // [B][L][D] f32  (or null)
    __bf16* __restrict__ CoutBf,    // [B][L][D] bf16 (or null)
    float*  __restrict__ Aout,      // [B][L][S] f32  (or null)
    long qbstride) {
    extern __shared__ char smem[];
    __bf16* Es = (__bf16*)smem;     // [16][SS]  = 64 KB

    const int b    = blockIdx.y;
    const int l0   = blockIdx.x * 16;
    const int tid  = threadIdx.x;
    const int wave = tid >> 5;
    const int lane = tid & 31;
    const int m    = lane & 15;
    const int kh   = lane >> 4;

    const __bf16* __restrict__ Qb = Q  + (size_t)b * qbstride;
    const __bf16* __restrict__ Kb = Kc + (size_t)b * SS * DD;
    const __bf16* __restrict__ Vb = Vt + (size_t)b * DD * SS;

    // ---- Phase 1: E tile [16 x S] ; wave w covers s in [w*256, w*256+256) ----
    // Q A-fragments are s-tile invariant: load the whole 16x512 row set once
    // into registers (16 frags x 8 VGPRs = 128 VGPRs) and reuse for 16 tiles.
    {
        v16bf qf[DD / 32];
        const __bf16* qrow = Qb + (size_t)(l0 + m) * DD;
#pragma unroll
        for (int kc = 0; kc < DD / 32; ++kc)
            qf[kc] = a_frag_bf16(qrow + kc * 32, kh);

        for (int j = 0; j < 16; ++j) {
            const int s0 = (wave * 16 + j) * 16;
            if (j < 15)  // prefetch next K s-tile row for this lane
                __builtin_prefetch(Kb + (size_t)(s0 + 16 + m) * DD, 0, 0);
            v8f acc = {};
#pragma unroll
            for (int kc = 0; kc < DD / 32; ++kc) {
                // B[k][n] = K[s0+n][kc*32+k] : lane n reads contiguous d
                v16bf bf = *(const v16bf*)(Kb + (size_t)(s0 + m) * DD +
                                           kc * 32 + kh * 16);
                acc = wmma_bf16(qf[kc], bf, acc);
            }
#pragma unroll
            for (int r = 0; r < 8; ++r) {
                int mm = r + (kh << 3);
                Es[mm * SS + s0 + m] = (__bf16)(acc[r] * SCALE);
            }
        }
    }
    __syncthreads();

    // ---- Phase 2: row softmax; half-wave (16 lanes) per row, shfl reduce ----
    {
        const int row = wave * 2 + kh;   // 8 waves * 2 = 16 rows
        const int t   = m;
        __bf16* erow = Es + row * SS;

        float mx = -3.0e38f;
        for (int i = t; i < SS; i += 16) mx = fmaxf(mx, (float)erow[i]);
#pragma unroll
        for (int off = 1; off < 16; off <<= 1)
            mx = fmaxf(mx, __shfl_xor(mx, off, 16));

        float sum = 0.0f;
        for (int i = t; i < SS; i += 16) {
            float p = __expf((float)erow[i] - mx);
            sum += p;
            erow[i] = (__bf16)p;
        }
#pragma unroll
        for (int off = 1; off < 16; off <<= 1)
            sum += __shfl_xor(sum, off, 16);
        const float inv = 1.0f / sum;

        if (Aout != nullptr) {
            float* arow = Aout + ((size_t)b * LL + l0 + row) * SS;
            for (int i = t; i < SS; i += 16) {
                float p = (float)erow[i] * inv;
                erow[i] = (__bf16)p;
                arow[i] = p;
            }
        } else {
            for (int i = t; i < SS; i += 16)
                erow[i] = (__bf16)((float)erow[i] * inv);
        }
    }
    __syncthreads();

    // ---- Phase 3: C[16 x D] = P[16 x S] . V[S x D] ----
    // Outer loop over k (s0) so one LDS A-frag feeds the wave's 4 d-tiles.
    {
        v8f acc[4] = {v8f{}, v8f{}, v8f{}, v8f{}};
        for (int s0 = 0; s0 < SS; s0 += 32) {
            v16bf a = a_frag_bf16(Es + m * SS + s0, kh);       // P from LDS
#pragma unroll
            for (int j = 0; j < 4; ++j) {
                const int d0 = (wave * 4 + j) * 16;
                // B[k][n] = V[s0+k][d0+n] = Vt[d0+n][s0+k] : contiguous/lane
                v16bf bf = *(const v16bf*)(Vb + (size_t)(d0 + m) * SS +
                                           s0 + kh * 16);
                acc[j] = wmma_bf16(a, bf, acc[j]);
            }
        }
#pragma unroll
        for (int j = 0; j < 4; ++j) {
            const int d0 = (wave * 4 + j) * 16;
#pragma unroll
            for (int r = 0; r < 8; ++r) {
                int mm = r + (kh << 3);
                size_t idx = ((size_t)b * LL + l0 + mm) * DD + d0 + m;
                if (Cout   != nullptr) Cout[idx]   = acc[j][r];
                if (CoutBf != nullptr) CoutBf[idx] = (__bf16)acc[j][r];
            }
        }
    }
}

// ---------------------------------------------------------------------------
extern "C" void kernel_launch(void* const* d_in, const int* in_sizes, int n_in,
                              void* d_out, int out_size, void* d_ws, size_t ws_size,
                              hipStream_t stream) {
    const float* H   = (const float*)d_in[0];   // [B,D,S]
    const float* K_w = (const float*)d_in[1];   // [D,D]
    const float* K_b = (const float*)d_in[2];   // [D]
    const float* V_w = (const float*)d_in[3];   // [D,D]
    const float* V_b = (const float*)d_in[4];   // [D]
    const float* Q_w = (const float*)d_in[5];   // [L,D]

    float* C_out = (float*)d_out;                         // [B,L,D]
    float* A_out = (float*)d_out + (size_t)BB * LL * DD;  // [B,L,S]

    // workspace layout (bf16)
    char* ws = (char*)d_ws;
    __bf16* Kbf  = (__bf16*)ws;                                     // [B,S,D]
    __bf16* Vbf  = (__bf16*)(ws + (size_t)BB * SS * DD * 2);        // [B,D,S]
    __bf16* Qbf  = (__bf16*)(ws + (size_t)2 * BB * SS * DD * 2);    // [L,D]
    __bf16* C1bf = (__bf16*)(ws + (size_t)2 * BB * SS * DD * 2
                                + (size_t)LL * DD * 2);             // [B,L,D]

    (void)in_sizes; (void)n_in; (void)out_size; (void)ws_size;

    // allow 64 KB dynamic LDS for attention kernel (idempotent, non-stream op)
    (void)hipFuncSetAttribute(reinterpret_cast<const void*>(attn_kernel),
                              hipFuncAttributeMaxDynamicSharedMemorySize, 65536);

    // 1) Q_w -> bf16
    {
        int n4 = (LL * DD) / 4;
        cvt_f32_bf16_kernel<<<(n4 + 255) / 256, 256, 0, stream>>>(Q_w, Qbf, n4);
    }

    // 2) K = elu(K_w . H + b)  -> [B,S,D] bf16 ; V -> [B,D,S] bf16 (transposed)
    {
        dim3 grid(DD / 64, SS / 16, BB);
        proj_elu_kernel<<<grid, 128, 0, stream>>>(K_w, K_b, H, Kbf, 1, DD);
        proj_elu_kernel<<<grid, 128, 0, stream>>>(V_w, V_b, H, Vbf, SS, 1);
    }

    // 3) pass 1: queries = Q_w (shared across batch) -> C1 (bf16 only)
    {
        dim3 grid(LL / 16, BB);
        attn_kernel<<<grid, 256, 65536, stream>>>(
            Qbf, Kbf, Vbf, nullptr, C1bf, nullptr, /*qbstride=*/0L);
    }

    // 4) pass 2: queries = C1 -> C (f32) and A (f32)
    {
        dim3 grid(LL / 16, BB);
        attn_kernel<<<grid, 256, 65536, stream>>>(
            C1bf, Kbf, Vbf, C_out, nullptr, A_out,
            /*qbstride=*/(long)LL * DD);
    }
}
